// AE_89343909691943
// MI455X (gfx1250) — matrix operand order repaired and checked by reading, not compiled, not run on hardware
//
#include <hip/hip_runtime.h>
#include <hip/hip_bf16.h>
#include <math.h>

// ---------------------------------------------------------------------------
// MI455X (gfx1250, wave32) implementation.
// All matmuls use V_WMMA_F32_16X16X32_BF16 (f32 accumulate); fp32 operands are
// converted to bf16 (RNE) while staging into LDS. The dominant 262-GFLOP gram
// matrices are fused with the loss reduction so no 4096x4096 matrix hits HBM.
// Intermediate activations are padded to 32-wide leading dims (zero-filled) so
// the A-side staging path has no bounds checks at all. The gram kernel stages
// its (already-bf16) tiles with CDNA5 async global->LDS copies (ASYNCcnt).
// ---------------------------------------------------------------------------

typedef __attribute__((ext_vector_type(16))) __bf16 v16bf;
typedef __attribute__((ext_vector_type(8)))  __bf16 v8bf;
typedef __attribute__((ext_vector_type(8)))  float  v8f;
typedef __attribute__((ext_vector_type(4)))  int    v4i;

#define STR 40   // LDS row stride in bf16 elems (80B rows keep 16B alignment)

#if defined(__HIP_DEVICE_COMPILE__) && __has_builtin(__builtin_amdgcn_global_load_async_to_lds_b128)
#define HAVE_ASYNC_LDS 1
#endif

__device__ inline __bf16 f2bf(float f) {
  union { float f; unsigned u; } a; a.f = f;
  unsigned u = a.u + 0x7fffu + ((a.u >> 16) & 1u);   // RNE
  union { unsigned short s; __bf16 b; } r;
  r.s = (unsigned short)(u >> 16);
  return r.b;
}

// pack two fp32 -> bf16x2 dword (lo in [15:0], hi in [31:16]), RNE
__device__ inline unsigned pk2(float a, float b) {
  union { float f; unsigned u; } x, y; x.f = a; y.f = b;
  unsigned ua = x.u + 0x7fffu + ((x.u >> 16) & 1u);
  unsigned ub = y.u + 0x7fffu + ((y.u >> 16) & 1u);
  return (ua >> 16) | (ub & 0xffff0000u);
}

// A fragment (16x32 bf16 tile, row-major in LDS, stride STR).
// CDNA5 layout: lanes<16 hold K 0..7 / 16..23, lanes>=16 hold 8..15 / 24..31.
__device__ inline v16bf load_a_frag(const __bf16* base, int lane) {
  int r  = lane & 15;
  int kb = (lane >> 4) << 3;
  const __bf16* p = base + r * STR + kb;
  union { v16bf v; v8bf h[2]; } u;
  u.h[0] = *(const v8bf*)(p);
  u.h[1] = *(const v8bf*)(p + 16);
  return u.v;
}

// B fragment (32x16 bf16, stored transposed [n][k] in LDS, stride STR).
// Lane layout: col = lane&15; lanes<16 hold K 0..7 / 16..23, lanes>=16 hold 8..15 / 24..31.
__device__ inline v16bf load_b_frag(const __bf16* base, int lane) {
  int n  = lane & 15;
  int kb = (lane >> 4) << 3;
  const __bf16* p = base + n * STR + kb;
  union { v16bf v; v8bf h[2]; } u;
  u.h[0] = *(const v8bf*)(p);
  u.h[1] = *(const v8bf*)(p + 16);
  return u.v;
}

// ---------------------------------------------------------------------------
// WMMA GEMM: C = op(A) @ B + bias.
// A: 4096 x lda fp32, lda % 32 == 0, fully valid (padded with zeros) -> the
//    K loop runs to Kpad (== multiple of 32) with NO guards on the A side.
// B: Kb x N row-major fp32; guarded only on the true K/N edges.
// C: 4096 x ldc fp32; pad columns [N, ldc) are zero-filled.
// Block: 256 thr = 8 waves; block tile 128x128; wave tile 32x64 (2x4 WMMA).
// ---------------------------------------------------------------------------
__global__ __launch_bounds__(256) void k_gemm(
    const float* __restrict__ A, int lda,
    const float* __restrict__ B, const float* __restrict__ bias,
    float* __restrict__ C, int ldc,
    int N, int Kpad, int Kb, int reluA, int sigOut)
{
  __shared__ __bf16 sA[128 * STR];
  __shared__ __bf16 sB[128 * STR];
  const int tid  = threadIdx.x;
  const int lane = tid & 31;
  const int wave = tid >> 5;
  const int m0 = blockIdx.y * 128;
  const int n0 = blockIdx.x * 128;
  const int wm = (wave & 3) * 32;
  const int wn = (wave >> 2) * 64;

  const int row = tid >> 1;            // staging: 2 threads per row/col
  const int ks  = (tid & 1) * 16;
  const int gn  = n0 + row;            // B staging column

  v8f acc[2][4];
  #pragma unroll
  for (int i = 0; i < 2; ++i)
    #pragma unroll
    for (int j = 0; j < 4; ++j)
      acc[i][j] = (v8f)(0.0f);

  for (int kt = 0; kt < Kpad; kt += 32) {
    { // ---- stage A (128x32): always a full tile, vectorized
      const float4* ap = (const float4*)(A + (size_t)(m0 + row) * lda + kt + ks);
      float4 f0 = ap[0], f1 = ap[1], f2 = ap[2], f3 = ap[3];
      if (kt + 32 < Kpad) __builtin_prefetch((const float*)ap + 32, 0, 1);
      if (reluA) {
        f0.x = fmaxf(f0.x, 0.f); f0.y = fmaxf(f0.y, 0.f); f0.z = fmaxf(f0.z, 0.f); f0.w = fmaxf(f0.w, 0.f);
        f1.x = fmaxf(f1.x, 0.f); f1.y = fmaxf(f1.y, 0.f); f1.z = fmaxf(f1.z, 0.f); f1.w = fmaxf(f1.w, 0.f);
        f2.x = fmaxf(f2.x, 0.f); f2.y = fmaxf(f2.y, 0.f); f2.z = fmaxf(f2.z, 0.f); f2.w = fmaxf(f2.w, 0.f);
        f3.x = fmaxf(f3.x, 0.f); f3.y = fmaxf(f3.y, 0.f); f3.z = fmaxf(f3.z, 0.f); f3.w = fmaxf(f3.w, 0.f);
      }
      uint4* d = (uint4*)(sA + row * STR + ks);
      d[0] = make_uint4(pk2(f0.x, f0.y), pk2(f0.z, f0.w), pk2(f1.x, f1.y), pk2(f1.z, f1.w));
      d[1] = make_uint4(pk2(f2.x, f2.y), pk2(f2.z, f2.w), pk2(f3.x, f3.y), pk2(f3.z, f3.w));
    }
    { // ---- stage B (32x128) transposed -> sB[n][k]; thread owns one column
      unsigned q[8];
      if (gn < N && kt + ks + 16 <= Kb) {       // interior fast path
        const float* bp = B + (size_t)(kt + ks) * N + gn;
        if (kt + ks + 48 <= Kb) __builtin_prefetch(bp + (size_t)32 * N, 0, 1);
        #pragma unroll
        for (int i = 0; i < 8; ++i)
          q[i] = pk2(bp[(size_t)(2 * i) * N], bp[(size_t)(2 * i + 1) * N]);
      } else {                                   // K/N edge
        #pragma unroll
        for (int i = 0; i < 8; ++i) {
          int k0 = kt + ks + 2 * i;
          float a = (gn < N && k0     < Kb) ? B[(size_t)k0 * N + gn]       : 0.0f;
          float b = (gn < N && k0 + 1 < Kb) ? B[(size_t)(k0 + 1) * N + gn] : 0.0f;
          q[i] = pk2(a, b);
        }
      }
      uint4* d = (uint4*)(sB + row * STR + ks);
      d[0] = make_uint4(q[0], q[1], q[2], q[3]);
      d[1] = make_uint4(q[4], q[5], q[6], q[7]);
    }
    __syncthreads();

    v16bf afr[2], bfr[4];
    #pragma unroll
    for (int i = 0; i < 2; ++i) afr[i] = load_a_frag(sA + (wm + i * 16) * STR, lane);
    #pragma unroll
    for (int j = 0; j < 4; ++j) bfr[j] = load_b_frag(sB + (wn + j * 16) * STR, lane);
    #pragma unroll
    for (int i = 0; i < 2; ++i)
      #pragma unroll
      for (int j = 0; j < 4; ++j)
        acc[i][j] = __builtin_amdgcn_wmma_f32_16x16x32_bf16(
            false, afr[i], false, bfr[j], (short)0, acc[i][j], false, false);
    __syncthreads();
  }

  // epilogue: rows always in range (M == 4096, exact grid); one col branch per tile
  #pragma unroll
  for (int i = 0; i < 2; ++i)
    #pragma unroll
    for (int j = 0; j < 4; ++j) {
      union { v8f v; float f[8]; } cu; cu.v = acc[i][j];
      int col   = n0 + wn + j * 16 + (lane & 15);
      int rbase = m0 + wm + i * 16 + ((lane >> 4) << 3);
      if (col < N) {
        float bv = bias[col];
        #pragma unroll
        for (int r = 0; r < 8; ++r) {
          float v = cu.f[r] + bv;
          if (sigOut) v = 1.0f / (1.0f + __expf(-v));
          C[(size_t)(rbase + r) * ldc + col] = v;
        }
      } else if (col < ldc) {                    // zero-fill K padding for next GEMM
        #pragma unroll
        for (int r = 0; r < 8; ++r)
          C[(size_t)(rbase + r) * ldc + col] = 0.0f;
      }
    }
}

// copy 16-wide bf16 row segments from row-major bf16 global into padded LDS
__device__ inline void fill_rows_bf(__bf16* dst, const __bf16* __restrict__ src,
                                    int ldsrc, int rowbase, int kt, int tid) {
  int row = tid >> 1;
  int ks  = (tid & 1) * 16;
  const __bf16* p = src + (size_t)(rowbase + row) * ldsrc + kt + ks;
  __bf16* d = dst + row * STR + ks;
#ifdef HAVE_ASYNC_LDS
  // CDNA5 async copy: global -> LDS without a VGPR round trip (ASYNCcnt).
  // builtin signature (from toolchain): (v4i* global_src, v4i* lds_dst, imm off, imm cpol)
  __builtin_amdgcn_global_load_async_to_lds_b128((v4i*)p,       (v4i*)d,       0, 0);
  __builtin_amdgcn_global_load_async_to_lds_b128((v4i*)(p + 8), (v4i*)(d + 8), 0, 0);
#else
  v8bf a = *(const v8bf*)(p);
  v8bf b = *(const v8bf*)(p + 8);
  *(v8bf*)(d)     = a;
  *(v8bf*)(d + 8) = b;
#endif
}

__device__ inline void wait_fill() {
#ifdef HAVE_ASYNC_LDS
#if __has_builtin(__builtin_amdgcn_s_wait_asynccnt)
  __builtin_amdgcn_s_wait_asynccnt(0);
#else
  asm volatile("s_wait_asynccnt 0x0" ::: "memory");
#endif
#endif
  __syncthreads();
}

// ---------------------------------------------------------------------------
// Fused gram + loss for one view:
//   S_tile = xn[I,:]@xn[J,:]^T - zn[I,:]@zn[J,:]^T   (second term via -zn copy)
//   loss += sum( (w[I] * S * w[J])^2 ) * scale
// 4096 % 128 == 0 and every K % 32 == 0 -> no bounds checks anywhere.
// ---------------------------------------------------------------------------
__global__ __launch_bounds__(256) void k_gramloss(
    const __bf16* __restrict__ Xn, int Kx,
    const __bf16* __restrict__ Znm,  // -zn  bf16 4096x128
    const __bf16* __restrict__ Zn,   // +zn  bf16 4096x128
    const float* __restrict__ wS, int v, float scale,
    float* __restrict__ lossAcc)
{
  __shared__ __bf16 sA[128 * STR];
  __shared__ __bf16 sB[128 * STR];
  const int tid  = threadIdx.x;
  const int lane = tid & 31;
  const int wave = tid >> 5;
  const int m0 = blockIdx.y * 128;
  const int n0 = blockIdx.x * 128;
  const int wm = (wave & 3) * 32;
  const int wn = (wave >> 2) * 64;

  v8f acc[2][4];
  #pragma unroll
  for (int i = 0; i < 2; ++i)
    #pragma unroll
    for (int j = 0; j < 4; ++j)
      acc[i][j] = (v8f)(0.0f);

  // phase 1: += xn_I . xn_J ; phase 2: += (-zn_I) . zn_J  == -= sim2
  for (int ph = 0; ph < 2; ++ph) {
    const __bf16* SA = ph ? Znm : Xn;
    const __bf16* SB = ph ? Zn  : Xn;
    const int     Kc = ph ? 128 : Kx;
    const int     ld = ph ? 128 : Kx;
    for (int kt = 0; kt < Kc; kt += 32) {
      fill_rows_bf(sA, SA, ld, m0, kt, tid);
      fill_rows_bf(sB, SB, ld, n0, kt, tid);
      wait_fill();
      v16bf afr[2], bfr[4];
      #pragma unroll
      for (int i = 0; i < 2; ++i) afr[i] = load_a_frag(sA + (wm + i * 16) * STR, lane);
      #pragma unroll
      for (int j = 0; j < 4; ++j) bfr[j] = load_b_frag(sB + (wn + j * 16) * STR, lane);
      #pragma unroll
      for (int i = 0; i < 2; ++i)
        #pragma unroll
        for (int j = 0; j < 4; ++j)
          acc[i][j] = __builtin_amdgcn_wmma_f32_16x16x32_bf16(
              false, afr[i], false, bfr[j], (short)0, acc[i][j], false, false);
      __syncthreads();
    }
  }

  // epilogue: weighted squared residual, block reduce, one atomic per block
  float p = 0.0f;
  #pragma unroll
  for (int i = 0; i < 2; ++i)
    #pragma unroll
    for (int j = 0; j < 4; ++j) {
      union { v8f v; float f[8]; } cu; cu.v = acc[i][j];
      int col   = n0 + wn + j * 16 + (lane & 15);
      int rbase = m0 + wm + i * 16 + ((lane >> 4) << 3);
      float wc = wS[(size_t)col * 6 + v];
      #pragma unroll
      for (int r = 0; r < 8; ++r) {
        float wr = wS[(size_t)(rbase + r) * 6 + v];
        float t  = wr * cu.f[r] * wc;
        p += t * t;
      }
    }
  float* red = (float*)sA;
  red[tid] = p;
  __syncthreads();
  for (int st = 128; st > 0; st >>= 1) {
    if (tid < st) red[tid] += red[tid + st];
    __syncthreads();
  }
  if (tid == 0) atomicAdd(lossAcc, red[0] * scale);
}

// batch-norm over batch axis (4096 rows), one block per channel, in place
__global__ __launch_bounds__(256) void k_bn(
    float* __restrict__ zl, const float* __restrict__ gamma, const float* __restrict__ beta)
{
  __shared__ float r1[256], r2[256];
  int c = blockIdx.x, tid = threadIdx.x;
  float s = 0.0f, q = 0.0f;
  for (int r = tid; r < 4096; r += 256) {
    float x = zl[(size_t)r * 128 + c];
    s += x; q += x * x;
  }
  r1[tid] = s; r2[tid] = q;
  __syncthreads();
  for (int st = 128; st > 0; st >>= 1) {
    if (tid < st) { r1[tid] += r1[tid + st]; r2[tid] += r2[tid + st]; }
    __syncthreads();
  }
  float mean = r1[0] * (1.0f / 4096.0f);
  float var  = r2[0] * (1.0f / 4096.0f) - mean * mean;
  float g = gamma[c] * rsqrtf(var + 1e-5f);
  float b = beta[c];
  for (int r = tid; r < 4096; r += 256) {
    size_t idx = (size_t)r * 128 + c;
    zl[idx] = (zl[idx] - mean) * g + b;
  }
}

// per-row l2 normalize + bf16 convert (gram inputs)
__global__ __launch_bounds__(256) void k_normx(
    const float* __restrict__ X, int D, __bf16* __restrict__ Xn)
{
  __shared__ float red[256];
  int r = blockIdx.x, tid = threadIdx.x;
  const float* xr = X + (size_t)r * D;
  float s = 0.0f;
  for (int c = tid; c < D; c += 256) { float v = xr[c]; s += v * v; }
  red[tid] = s;
  __syncthreads();
  for (int st = 128; st > 0; st >>= 1) {
    if (tid < st) red[tid] += red[tid + st];
    __syncthreads();
  }
  float inv = 1.0f / fmaxf(sqrtf(red[0]), 1e-12f);
  __bf16* on = Xn + (size_t)r * D;
  for (int c = tid; c < D; c += 256) on[c] = f2bf(xr[c] * inv);
}

// weighted fusion z, we_s, +/- zn bf16. one block (128 thr) per row.
__global__ __launch_bounds__(128) void k_fusez(
    const float* __restrict__ we, const float* __restrict__ vars,
    const float* __restrict__ zs, float* __restrict__ z,
    float* __restrict__ weS, __bf16* __restrict__ ZnBf, __bf16* __restrict__ ZnmBf)
{
  __shared__ float sw[6];
  __shared__ float red[128];
  int r = blockIdx.x, tid = threadIdx.x;
  if (tid < 6) {
    float w = we[(size_t)r * 6 + tid] * vars[tid];
    sw[tid] = w;
    weS[(size_t)r * 6 + tid] = w;
  }
  __syncthreads();
  float sum = sw[0] + sw[1] + sw[2] + sw[3] + sw[4] + sw[5];
  float zv = 0.0f;
  #pragma unroll
  for (int i = 0; i < 6; ++i)
    zv += (sw[i] / sum) * zs[(size_t)i * 4096 * 128 + (size_t)r * 128 + tid];
  z[(size_t)r * 128 + tid] = zv;
  red[tid] = zv * zv;
  __syncthreads();
  for (int st = 64; st > 0; st >>= 1) {
    if (tid < st) red[tid] += red[tid + st];
    __syncthreads();
  }
  float inv = 1.0f / fmaxf(sqrtf(red[0]), 1e-12f);
  float zn  = zv * inv;
  ZnBf[(size_t)r * 128 + tid]  = f2bf(zn);
  ZnmBf[(size_t)r * 128 + tid] = f2bf(-zn);
}

__global__ void k_zero(float* p) {
  if (threadIdx.x == 0 && blockIdx.x == 0) *p = 0.0f;
}

// ---------------------------------------------------------------------------
extern "C" void kernel_launch(void* const* d_in, const int* in_sizes, int n_in,
                              void* d_out, int out_size, void* d_ws, size_t ws_size,
                              hipStream_t stream)
{
  (void)in_sizes; (void)n_in; (void)out_size; (void)ws_size;
  static const int ND[6]  = {1024, 1536, 2048, 1280, 896, 1024};
  static const int D0[6]  = { 819, 1229, 1638, 1024, 717,  819};
  static const int D0p[6] = { 832, 1248, 1664, 1024, 736,  832}; // pad32
  const int N = 4096, HID = 1500, HIDp = 1504, NZ = 128, NL = 80;

  // inputs: x0..x5, we, then params pytree in jax sorted-key order:
  //   enc[v]: W1, W2, Wz, b1, b2, beta, bz, gamma  (x6), then reg_W, reg_b, variables
  const float* x[6];
  for (int i = 0; i < 6; ++i) x[i] = (const float*)d_in[i];
  const float* we = (const float*)d_in[6];
  const float *W1[6], *W2[6], *Wz[6], *b1[6], *b2[6], *beta[6], *bz[6], *gamma[6];
  int pi = 7;
  for (int v = 0; v < 6; ++v) {
    W1[v]    = (const float*)d_in[pi++];
    W2[v]    = (const float*)d_in[pi++];
    Wz[v]    = (const float*)d_in[pi++];
    b1[v]    = (const float*)d_in[pi++];
    b2[v]    = (const float*)d_in[pi++];
    beta[v]  = (const float*)d_in[pi++];
    bz[v]    = (const float*)d_in[pi++];
    gamma[v] = (const float*)d_in[pi++];
  }
  const float* regW = (const float*)d_in[pi++];
  const float* regB = (const float*)d_in[pi++];
  const float* vars = (const float*)d_in[pi++];

  // output layout: yLable | z | individual_zs | y_specific | loss
  float* out  = (float*)d_out;
  float* yL   = out;
  float* zOut = out + (size_t)N * NL;
  float* zsO  = zOut + (size_t)N * NZ;
  float* ySp  = zsO + (size_t)6 * N * NZ;
  float* loss = ySp + (size_t)6 * N * NL;

  // workspace (~70.9 MB): h1(4096x1664) | h2(4096x1504) | xn_bf | zn_bf | znm_bf | we_s
  char* ws = (char*)d_ws;
  float*  h1    = (float*)ws;
  float*  h2    = (float*)(ws + (size_t)27262976);
  __bf16* xnbf  = (__bf16*)(ws + (size_t)51904512);
  __bf16* znbf  = (__bf16*)(ws + (size_t)68681728);
  __bf16* znmbf = (__bf16*)(ws + (size_t)69730304);
  float*  weS   = (float*)(ws + (size_t)70778880);

  // encoders (per view); BN in place on the individual_zs slice of d_out
  for (int v = 0; v < 6; ++v) {
    float* zl = zsO + (size_t)v * N * NZ;
    k_gemm<<<dim3((D0[v] + 127) / 128, 32), 256, 0, stream>>>(
        x[v], ND[v], W1[v], b1[v], h1, D0p[v], D0[v], ND[v], ND[v], 0, 0);
    k_gemm<<<dim3((HID + 127) / 128, 32), 256, 0, stream>>>(
        h1, D0p[v], W2[v], b2[v], h2, HIDp, HID, D0p[v], D0[v], 1, 0);
    k_gemm<<<dim3(1, 32), 256, 0, stream>>>(
        h2, HIDp, Wz[v], bz[v], zl, NZ, NZ, HIDp, HID, 1, 0);
    k_bn<<<128, 256, 0, stream>>>(zl, gamma[v], beta[v]);
  }

  // fusion: z, we_s, +/- zn (bf16)
  k_fusez<<<4096, 128, 0, stream>>>(we, vars, zsO, zOut, weS, znbf, znmbf);

  // fused gram + loss, one view at a time (xnbf reused)
  k_zero<<<1, 32, 0, stream>>>(loss);
  const float scale = 1.0f / (6.0f * 4096.0f * 4096.0f);
  for (int v = 0; v < 6; ++v) {
    k_normx<<<4096, 256, 0, stream>>>(x[v], ND[v], xnbf);
    k_gramloss<<<dim3(32, 32), 256, 0, stream>>>(
        xnbf, ND[v], znmbf, znbf, weS, v, scale, loss);
  }

  // reg heads: sigmoid(relu(h) @ reg_W + reg_b)
  k_gemm<<<dim3(1, 32), 256, 0, stream>>>(zOut, NZ, regW, regB, yL, NL, NL, NZ, NZ, 1, 1);
  for (int v = 0; v < 6; ++v)
    k_gemm<<<dim3(1, 32), 256, 0, stream>>>(
        zsO + (size_t)v * N * NZ, NZ, regW, regB, ySp + (size_t)v * N * NL, NL, NL, NZ, NZ, 1, 1);
}
